// PointAttentionBlock_70987219468464
// MI455X (gfx1250) — matrix operand order repaired and checked by reading, not compiled, not run on hardware
//
#include <hip/hip_runtime.h>

typedef __attribute__((ext_vector_type(16))) _Float16     v16h;
typedef __attribute__((ext_vector_type(8)))  float        v8f;
typedef __attribute__((ext_vector_type(4)))  unsigned int u32x4;
typedef __attribute__((ext_vector_type(8)))  int          i32x8;
typedef __attribute__((ext_vector_type(4)))  int          i32x4;

#define N_PTS 1024
#define M_PTS 1024
#define DD    32
#define CIN   64
#define MASKVAL (-3.402823466e+38f)

// ---------------------------------------------------------------------------
// Kernel 1: q = Wq*h_query + bq (32x1024), k = Wkv[0:32]*h_ref + bkv[0:32].
// (v rows of Wkv are dead: the reference returns attn only.)
// ---------------------------------------------------------------------------
__global__ __launch_bounds__(256) void qk_precompute(
    const float* __restrict__ h_query, const float* __restrict__ h_ref,
    const float* __restrict__ Wq,  const float* __restrict__ bq,
    const float* __restrict__ Wkv, const float* __restrict__ bkv,
    float* __restrict__ qbuf, float* __restrict__ kbuf) {
  int i   = blockIdx.x * blockDim.x + threadIdx.x;   // 0..65535
  int sel = i >> 15;
  int idx = i & 32767;
  int d = idx >> 10, p = idx & 1023;
  if (sel == 0) {
    float acc = bq[d];
#pragma unroll 8
    for (int c = 0; c < CIN; ++c)
      acc = fmaf(Wq[d * CIN + c], h_query[c * N_PTS + p], acc);
    qbuf[d * N_PTS + p] = acc;
  } else {
    float acc = bkv[d];
#pragma unroll 8
    for (int c = 0; c < CIN; ++c)
      acc = fmaf(Wkv[d * CIN + c], h_ref[c * M_PTS + p], acc);
    kbuf[d * M_PTS + p] = acc;
  }
}

// One v_cvt_pk_rtz_f16_f32: packs two f32 into an f16x2 dword.
__device__ inline int pk2(float a, float b) {
  return __builtin_bit_cast(int, __builtin_amdgcn_cvt_pkrtz(a, b));
}

union BV { v16h v; int i[8]; };

// f16 A-fragment (16x32) for W[row, K], contiguous-K layout:
// lanes 0-15 row M=lane, K in {0..15}; lanes 16-31 same rows, K {16..31}.
__device__ inline v16h load_a_frag(const float* __restrict__ W, int lane, int tile) {
  int row = (lane & 15) + 16 * tile;
  int hlf = lane >> 4;
  BV a;
#pragma unroll
  for (int w = 0; w < 8; ++w) {
    int j0 = 2 * w;                                      // element pair j0, j0+1
    int K0 = hlf * 16 + j0;
    a.i[w] = pk2(W[row * DD + K0], W[row * DD + K0 + 1]);
  }
  return a.v;
}

// Convert D-fragment pair (rows 0..15 in d0, 16..31 in d1) to a B-fragment
// (lane half selects K 0..15 / 16..31). Pack to f16 dwords FIRST (8x
// v_cvt_pk_rtz_f16_f32), then one xor-16 shuffle per packed dword:
// 8 ds_bpermute + 8 cndmask per conversion.
__device__ inline v16h dpair_to_b(const v8f d0, const v8f d1, int hlf) {
  int p0[4], p1[4];
#pragma unroll
  for (int w = 0; w < 4; ++w) {
    p0[w] = pk2(d0[2 * w], d0[2 * w + 1]);
    p1[w] = pk2(d1[2 * w], d1[2 * w + 1]);
  }
  BV bv;
#pragma unroll
  for (int w = 0; w < 4; ++w) {
    int o0 = __shfl_xor(p0[w], 16, 32);
    int o1 = __shfl_xor(p1[w], 16, 32);
    bv.i[w]     = hlf ? o1 : p0[w];   // K = hlf*16 + {2w,2w+1}
    bv.i[w + 4] = hlf ? p1[w] : o0;   // K = hlf*16 + 8 + {2w,2w+1}
  }
  return bv.v;
}

// ---------------------------------------------------------------------------
// Kernel 2: one workgroup per n-row. Wave 0 DMAs the rel_pos row (2D tile
// 1024x3, line stride N*M) into LDS via the Tensor Data Mover while the block
// stages weights; 8 waves then sweep the 64 m-tiles with chained
// v_wmma_f32_16x16x32_f16; sim[32][1024] lives in LDS (128 KB of the 320 KB
// WGP pool) and is softmaxed in place, so HBM traffic ~= the 128 MB output.
// ---------------------------------------------------------------------------
__global__ __launch_bounds__(256) void attn_kernel(
    const float* __restrict__ rel_pos, const unsigned char* __restrict__ mask,
    const float* __restrict__ Wp1, const float* __restrict__ bp1,
    const float* __restrict__ Wp2, const float* __restrict__ bp2,
    const float* __restrict__ Wa1, const float* __restrict__ ba1,
    const float* __restrict__ Wa2, const float* __restrict__ ba2,
    const float* __restrict__ qbuf, const float* __restrict__ kbuf,
    float* __restrict__ out) {
  extern __shared__ float smem[];
  float* s_sim = smem;                   // 32*1024 floats = 128 KB
  float* s_rel = smem + DD * M_PTS;      // 3*1024 floats  = 12 KB (TDM target)
  float* s_wp1 = s_rel + 3 * M_PTS;      // 96
  float* s_bp1 = s_wp1 + 96;             // 32
  float* s_bp2 = s_bp1 + 32;             // 32
  float* s_ba1 = s_bp2 + 32;             // 32
  float* s_ba2 = s_ba1 + 32;             // 32
  float* s_q   = s_ba2 + 32;             // 32  (q[:, n])

  const int n    = blockIdx.x;
  const int tid  = threadIdx.x;
  const int lane = tid & 31;
  const int wave = tid >> 5;
  const int nwaves = blockDim.x >> 5;
  const int col  = lane & 15;
  const int hlf  = lane >> 4;

  // ---- TDM: DMA rel_pos[0:3, n, 0:1024] into s_rel (issued by wave 0) ----
  if (wave == 0) {
    unsigned long long ga = (unsigned long long)(uintptr_t)(rel_pos + (size_t)n * M_PTS);
    unsigned int lds_off  = (unsigned int)(uintptr_t)s_rel;   // low 32 bits = LDS offset
    u32x4 g0;
    g0[0] = 1u;                                        // count=1 (valid descriptor)
    g0[1] = lds_off;                                   // lds_addr
    g0[2] = (unsigned int)ga;                          // global_addr[31:0]
    g0[3] = (unsigned int)((ga >> 32) & 0x01FFFFFFull) // global_addr[56:32]
            | 0x80000000u;                             // type=2 ("image")
    i32x8 g1;
    g1[0] = (2 << 16);            // data_size=4B; no multicast/pad/iterate
    g1[1] = (1024 << 16);         // tensor_dim0[15:0]=1024
    g1[2] = (3 << 16);            // tensor_dim0 hi=0 | tensor_dim1=3
    g1[3] = (1024 << 16);         // tensor_dim1 hi=0 | tile_dim0=1024
    g1[4] = 3;                    // tile_dim1=3 | tile_dim2=0
    g1[5] = (int)(N_PTS * M_PTS); // tensor_dim0_stride[31:0] = 2^20 elements
    g1[6] = 0;                    // stride0 hi | tensor_dim1_stride lo
    g1[7] = 0;
    i32x4 z4 = {0, 0, 0, 0};
#if __clang_major__ >= 23
    i32x8 z8 = {0, 0, 0, 0, 0, 0, 0, 0};
    __builtin_amdgcn_tensor_load_to_lds(g0, g1, z4, z4, z8, 0);
#else
    __builtin_amdgcn_tensor_load_to_lds(g0, g1, z4, z4, 0);
#endif
  }

  // ---- cooperative staging of the small weight arrays ----
  if (tid < 96) s_wp1[tid] = Wp1[tid];
  if (tid < 32) {
    s_bp1[tid] = bp1[tid];
    s_bp2[tid] = bp2[tid];
    s_ba1[tid] = ba1[tid];
    s_ba2[tid] = ba2[tid];
    s_q[tid]   = qbuf[tid * N_PTS + n];
  }

  // Weight A-fragments, resident in VGPRs for the whole kernel.
  v16h aP_lo = load_a_frag(Wp2, lane, 0);
  v16h aP_hi = load_a_frag(Wp2, lane, 1);
  v16h a1_lo = load_a_frag(Wa1, lane, 0);
  v16h a1_hi = load_a_frag(Wa1, lane, 1);
  v16h a2_lo = load_a_frag(Wa2, lane, 0);
  v16h a2_hi = load_a_frag(Wa2, lane, 1);

  __syncthreads();

  // ---- hoist all loop-invariant per-lane values out of LDS into VGPRs ----
  float wx[16], wy[16], wz[16], bb1[16];
#pragma unroll
  for (int j = 0; j < 16; ++j) {
    int c = hlf * 16 + j;
    wx[j]  = s_wp1[c * 3 + 0];
    wy[j]  = s_wp1[c * 3 + 1];
    wz[j]  = s_wp1[c * 3 + 2];
    bb1[j] = s_bp1[c];
  }
  float qb0[8], qb1[8], A1b0[8], A1b1[8], A2b0[8], A2b1[8];
#pragma unroll
  for (int r = 0; r < 8; ++r) {
    int cc0 = r + hlf * 8, cc1 = cc0 + 16;
    qb0[r]  = s_q[cc0] + s_bp2[cc0];
    qb1[r]  = s_q[cc1] + s_bp2[cc1];
    A1b0[r] = s_ba1[cc0];  A1b1[r] = s_ba1[cc1];
    A2b0[r] = s_ba2[cc0];  A2b1[r] = s_ba2[cc1];
  }

  if (wave == 0) __builtin_amdgcn_s_wait_tensorcnt(0);  // rel_pos tile landed
  __syncthreads();

  for (int t = wave; t < M_PTS / 16; t += nwaves) {
    const int m = t * 16 + col;
    if (t + nwaves < M_PTS / 16)                        // next k panel (L2 hit)
      __builtin_prefetch(&kbuf[(t + nwaves) * 16], 0, 3);

    // pos-MLP layer 1 (K=3) straight into B-fragment layout from LDS rel tile.
    float x = s_rel[0 * M_PTS + m];
    float y = s_rel[1 * M_PTS + m];
    float z = s_rel[2 * M_PTS + m];
    float pv[16];
#pragma unroll
    for (int j = 0; j < 16; ++j)
      pv[j] = fmaxf(fmaf(wx[j], x, fmaf(wy[j], y, fmaf(wz[j], z, bb1[j]))), 0.f);
    BV b1;
#pragma unroll
    for (int w = 0; w < 8; ++w) b1.i[w] = pk2(pv[2 * w], pv[2 * w + 1]);

    // Fold q - k + bp2 into the accumulator: D of the pos GEMM == qk_rel.
    v8f c0, c1;
#pragma unroll
    for (int r = 0; r < 8; ++r) {
      int cc0 = r + hlf * 8, cc1 = cc0 + 16;
      c0[r] = qb0[r] - kbuf[cc0 * M_PTS + m];
      c1[r] = qb1[r] - kbuf[cc1 * M_PTS + m];
    }
    v8f qk0 = __builtin_amdgcn_wmma_f32_16x16x32_f16(false, aP_lo, false, b1.v,
                                                     (short)0, c0, false, false);
    v8f qk1 = __builtin_amdgcn_wmma_f32_16x16x32_f16(false, aP_hi, false, b1.v,
                                                     (short)0, c1, false, false);

    // attn-MLP layer 1 (+ ReLU)
    v16h b2 = dpair_to_b(qk0, qk1, hlf);
#pragma unroll
    for (int r = 0; r < 8; ++r) { c0[r] = A1b0[r]; c1[r] = A1b1[r]; }
    v8f h0 = __builtin_amdgcn_wmma_f32_16x16x32_f16(false, a1_lo, false, b2,
                                                    (short)0, c0, false, false);
    v8f h1 = __builtin_amdgcn_wmma_f32_16x16x32_f16(false, a1_hi, false, b2,
                                                    (short)0, c1, false, false);
#pragma unroll
    for (int r = 0; r < 8; ++r) {
      h0[r] = fmaxf(h0[r], 0.f);
      h1[r] = fmaxf(h1[r], 0.f);
    }

    // attn-MLP layer 2 -> sim
    v16h b3 = dpair_to_b(h0, h1, hlf);
#pragma unroll
    for (int r = 0; r < 8; ++r) { c0[r] = A2b0[r]; c1[r] = A2b1[r]; }
    v8f s0 = __builtin_amdgcn_wmma_f32_16x16x32_f16(false, a2_lo, false, b3,
                                                    (short)0, c0, false, false);
    v8f s1 = __builtin_amdgcn_wmma_f32_16x16x32_f16(false, a2_hi, false, b3,
                                                    (short)0, c1, false, false);

    // Mask and park sim in LDS (row-major per d for the softmax sweep).
    const bool mk = mask[n * M_PTS + m] != 0;
#pragma unroll
    for (int r = 0; r < 8; ++r) {
      int cc0 = r + hlf * 8, cc1 = cc0 + 16;
      s_sim[cc0 * M_PTS + m] = mk ? s0[r] : MASKVAL;
      s_sim[cc1 * M_PTS + m] = mk ? s1[r] : MASKVAL;
    }
  }

  __syncthreads();

  // Masked softmax over m for each of the 32 d-rows; one wave per row.
  for (int c = wave; c < DD; c += nwaves) {
    float* row = s_sim + c * M_PTS;
    float mx = MASKVAL;
    for (int j = lane; j < M_PTS; j += 32) mx = fmaxf(mx, row[j]);
#pragma unroll
    for (int off = 16; off > 0; off >>= 1) mx = fmaxf(mx, __shfl_xor(mx, off, 32));
    float sum = 0.f;
    for (int j = lane; j < M_PTS; j += 32) {
      float e = __expf(row[j] - mx);
      row[j] = e;
      sum += e;
    }
#pragma unroll
    for (int off = 16; off > 0; off >>= 1) sum += __shfl_xor(sum, off, 32);
    const float inv = 1.f / sum;  // all-masked row -> sum==1024 -> uniform (matches ref)
    float4* orow = (float4*)(out + (size_t)c * N_PTS * M_PTS + (size_t)n * M_PTS);
    const float4* lrow = (const float4*)row;
    for (int j = lane; j < M_PTS / 4; j += 32) {
      float4 v = lrow[j];
      orow[j] = make_float4(v.x * inv, v.y * inv, v.z * inv, v.w * inv);
    }
  }
}

extern "C" void kernel_launch(void* const* d_in, const int* in_sizes, int n_in,
                              void* d_out, int out_size, void* d_ws, size_t ws_size,
                              hipStream_t stream) {
  const float*         h_query = (const float*)d_in[0];
  const float*         h_ref   = (const float*)d_in[1];
  const float*         rel_pos = (const float*)d_in[2];
  const unsigned char* mask    = (const unsigned char*)d_in[3]; // jnp.bool_ = 1 byte
  const float* Wq  = (const float*)d_in[4];
  const float* bq  = (const float*)d_in[5];
  const float* Wkv = (const float*)d_in[6];
  const float* bkv = (const float*)d_in[7];
  const float* Wp1 = (const float*)d_in[8];
  const float* bp1 = (const float*)d_in[9];
  const float* Wp2 = (const float*)d_in[10];
  const float* bp2 = (const float*)d_in[11];
  const float* Wa1 = (const float*)d_in[12];
  const float* ba1 = (const float*)d_in[13];
  const float* Wa2 = (const float*)d_in[14];
  const float* ba2 = (const float*)d_in[15];
  float* out  = (float*)d_out;
  float* qbuf = (float*)d_ws;             // 32*1024 floats
  float* kbuf = qbuf + DD * N_PTS;        // 32*1024 floats

  qk_precompute<<<256, 256, 0, stream>>>(h_query, h_ref, Wq, bq, Wkv, bkv,
                                         qbuf, kbuf);

  const size_t lds_bytes =
      (size_t)(DD * M_PTS + 3 * M_PTS + 96 + 5 * 32) * sizeof(float);
  attn_kernel<<<N_PTS, 256, lds_bytes, stream>>>(rel_pos, mask, Wp1, bp1, Wp2,
                                                 bp2, Wa1, ba1, Wa2, ba2, qbuf,
                                                 kbuf, out);
}
